// CustomEPA_42803644072204
// MI455X (gfx1250) — matrix operand order repaired and checked by reading, not compiled, not run on hardware
//
#include <hip/hip_runtime.h>
#include <hip/hip_bf16.h>

typedef _Float16 v16h __attribute__((ext_vector_type(16)));
typedef float    v8f  __attribute__((ext_vector_type(8)));

#define AS3 __attribute__((address_space(3)))

union Frag { uint4 u2[2]; v16h h; };
union BU  { _Float16 h[8]; uint4 u; };

#define B_    8
#define N_    4096
#define C_    768
#define H_    4
#define DH    192
#define QKVLD 2304   // 3*C

// ---------------------------------------------------------------- async helpers
__device__ __forceinline__ unsigned lds_off(const void* p) {
  return (unsigned)(size_t)(AS3 const char*)p;   // LDS byte offset (AS3 ptrs are 32-bit)
}
// GLOBAL_LOAD_ASYNC_TO_LDS_B128: VDST = LDS byte address, VADDR = 64-bit global addr
__device__ __forceinline__ void async_ld_b128(const _Float16* g, void* l) {
  asm volatile("global_load_async_to_lds_b128 %0, %1, off"
               :: "v"(lds_off(l)), "v"(g) : "memory");
}
__device__ __forceinline__ void async_wait0() {
  asm volatile("s_wait_asynccnt 0" ::: "memory");
}

// ---------------------------------------------------------------- conversions
__global__ void cvt_f32_f16(const float* __restrict__ src,
                            _Float16* __restrict__ dst, int n) {
  int i = blockIdx.x * blockDim.x + threadIdx.x;
  int stride = gridDim.x * blockDim.x;
  for (; i < n; i += stride) dst[i] = (_Float16)src[i];
}

// ---------------------------------------------------------------- generic GEMM
// C[M,N] = A[M,K] (row-major f16) @ B[K,N] (row-major f16); BM=128, BK=32.
// BN=128: 8 waves in 2x4, wave tile 64x32 (4x2 WMMA / iter).
// BN=64 : 8 waves in 4x2, wave tile 32x32 (2x2 WMMA / iter).
// Double-buffered pipeline: A tile via async global->LDS, B tile via reg prefetch
// + transposed ds_store. MODE 0: f32 out + bias ; MODE 1: f16 out.
template<int MODE, int BN>
__global__ __launch_bounds__(256) void gemm_rr(
    const _Float16* __restrict__ A, long long lda, long long sAb, long long sAh,
    const _Float16* __restrict__ Bm, long long ldb, long long sBb, long long sBh,
    void* __restrict__ Cm, long long ldc, long long sCb, long long sCh,
    const float* __restrict__ bias, int K)
{
  constexpr int WROWS = (BN == 128) ? 2 : 4;    // wave grid rows
  constexpr int WCOLS = 8 / WROWS;              // wave grid cols
  constexpr int MSUB  = (128 / WROWS) / 16;     // 4 or 2
  constexpr int NSUB  = (BN / WCOLS) / 16;      // 2
  constexpr int KPT   = BN / 8;                 // B k-elems per thread (16 or 8)

  __shared__ _Float16 lA[2][128 * 40];  // [m][k], stride 40 (80B: 16B aligned, conflict-free)
  __shared__ _Float16 lB[2][BN * 40];   // [n][k] (transposed on store)

  const int t = threadIdx.x;
  const int lane = t & 31, wave = t >> 5;
  const int l16 = lane & 15, halfl = lane >> 4;
  const int wm = wave / WCOLS, wn = wave % WCOLS;
  const long long zb = blockIdx.z >> 2, zh = blockIdx.z & 3;

  // per-thread source/dest positions for cooperative tile loads
  const _Float16* Ag = A + zb * sAb + zh * sAh +
                       ((long long)blockIdx.y * 128 + (t >> 1)) * lda + (t & 1) * 16;
  const int bn = t % BN;
  const int bk = (t / BN) * KPT;
  const _Float16* Bg = Bm + zb * sBb + zh * sBh + blockIdx.x * BN +
                       (long long)bk * ldb + bn;
  const int aoff = (t >> 1) * 40 + (t & 1) * 16;
  const int boff = bn * 40 + bk;

  v8f acc[MSUB][NSUB] = {};
  const int ntiles = K >> 5;

  // ---- prologue: stage tile 0 into buffer 0
  {
    async_ld_b128(Ag,     &lA[0][aoff]);
    async_ld_b128(Ag + 8, &lA[0][aoff + 8]);
    BU tb[KPT / 8];
#pragma unroll
    for (int c = 0; c < KPT / 8; ++c)
#pragma unroll
      for (int j = 0; j < 8; ++j) tb[c].h[j] = Bg[(long long)(c * 8 + j) * ldb];
#pragma unroll
    for (int c = 0; c < KPT / 8; ++c) *(uint4*)&lB[0][boff + c * 8] = tb[c].u;
    async_wait0();
  }
  __syncthreads();

  for (int i = 0; i < ntiles; ++i) {
    const int cur = i & 1, nxt = cur ^ 1;
    const bool more = (i + 1) < ntiles;
    BU tb[KPT / 8];
    if (more) {
      const _Float16* gA = Ag + (i + 1) * 32;
      const _Float16* gB = Bg + (long long)(i + 1) * 32 * ldb;
      async_ld_b128(gA,     &lA[nxt][aoff]);      // overlaps with WMMA below
      async_ld_b128(gA + 8, &lA[nxt][aoff + 8]);
#pragma unroll
      for (int c = 0; c < KPT / 8; ++c)
#pragma unroll
        for (int j = 0; j < 8; ++j) tb[c].h[j] = gB[(long long)(c * 8 + j) * ldb];
    }

    Frag af[MSUB], bf[NSUB];
#pragma unroll
    for (int ms = 0; ms < MSUB; ++ms) {
      int r = wm * (128 / WROWS) + ms * 16 + l16;
      af[ms].u2[0] = *(const uint4*)&lA[cur][r * 40 + halfl * 8];       // K = half*8+0..7
      af[ms].u2[1] = *(const uint4*)&lA[cur][r * 40 + 16 + halfl * 8];  // K = 16+half*8+0..7
    }
#pragma unroll
    for (int ns = 0; ns < NSUB; ++ns) {
      int c = wn * (BN / WCOLS) + ns * 16 + l16;
      bf[ns].u2[0] = *(const uint4*)&lB[cur][c * 40 + halfl * 8];       // K = half*8+0..7
      bf[ns].u2[1] = *(const uint4*)&lB[cur][c * 40 + 16 + halfl * 8];  // K = 16+half*8+0..7
    }
#pragma unroll
    for (int ms = 0; ms < MSUB; ++ms)
#pragma unroll
      for (int ns = 0; ns < NSUB; ++ns)
        acc[ms][ns] = __builtin_amdgcn_wmma_f32_16x16x32_f16(
            false, af[ms].h, false, bf[ns].h, (short)0, acc[ms][ns], false, false);

    if (more) {
#pragma unroll
      for (int c = 0; c < KPT / 8; ++c) *(uint4*)&lB[nxt][boff + c * 8] = tb[c].u;
      async_wait0();
    }
    __syncthreads();
  }

  const long long Cbase = zb * sCb + zh * sCh +
                          (long long)blockIdx.y * 128 * ldc + blockIdx.x * BN;
#pragma unroll
  for (int ms = 0; ms < MSUB; ++ms)
#pragma unroll
    for (int ns = 0; ns < NSUB; ++ns) {
      int gc = wn * (BN / WCOLS) + ns * 16 + l16;
#pragma unroll
      for (int r = 0; r < 8; ++r) {
        int gr = wm * (128 / WROWS) + ms * 16 + r + 8 * halfl;
        long long idx = Cbase + (long long)gr * ldc + gc;
        float v = acc[ms][ns][r];
        if constexpr (MODE == 0) {
          if (bias) v += bias[blockIdx.x * BN + gc];
          ((float*)Cm)[idx] = v;
        } else {
          ((_Float16*)Cm)[idx] = (_Float16)v;
        }
      }
    }
}

// ---------------------------------------------------------------- q.kT per (b,h)
// attn_raw[beta][d][e] = sum_n q[d,n]*k[e,n]; 64x64 tile, K=N_=4096.
// 8 waves in 2x4; wave tile 32x16 (2 WMMA). Double-buffered register prefetch.
__global__ __launch_bounds__(256) void attn_qk(
    const _Float16* __restrict__ qkv, float* __restrict__ attn_raw)
{
  __shared__ _Float16 lQ[2][64 * 40];   // [d][n]
  __shared__ _Float16 lK[2][64 * 40];   // [e][n]
  const int t = threadIdx.x;
  const int lane = t & 31, wave = t >> 5;
  const int l16 = lane & 15, halfl = lane >> 4;
  const int wm = wave >> 2, wn = wave & 3;
  const int bb = blockIdx.z >> 2, hh = blockIdx.z & 3;
  const int d0 = blockIdx.y * 64, e0 = blockIdx.x * 64;
  const long long rowbase = (long long)bb * N_ * QKVLD;
  const int qcol = hh * DH + d0;
  const int kcol = C_ + hh * DH + e0;

  v8f acc[2] = {};
  const int ld = t & 63;               // d (or e) within tile
  const int nb = (t >> 6) * 8;         // n sub-block
  const int soff = ld * 40 + nb;

  // prologue: tile 0 -> buffer 0
  {
    BU qv, kv;
#pragma unroll
    for (int i = 0; i < 8; ++i) {
      long long row = rowbase + (long long)(nb + i) * QKVLD;
      qv.h[i] = qkv[row + qcol + ld];
      kv.h[i] = qkv[row + kcol + ld];
    }
    *(uint4*)&lQ[0][soff] = qv.u;
    *(uint4*)&lK[0][soff] = kv.u;
  }
  __syncthreads();

  for (int it = 0; it < (N_ >> 5); ++it) {
    const int cur = it & 1, nxt = cur ^ 1;
    const bool more = (it + 1) < (N_ >> 5);
    BU qv, kv;
    if (more) {
#pragma unroll
      for (int i = 0; i < 8; ++i) {
        long long row = rowbase + (long long)((it + 1) * 32 + nb + i) * QKVLD;
        qv.h[i] = qkv[row + qcol + ld];
        kv.h[i] = qkv[row + kcol + ld];
      }
    }

    Frag af[2], bf;
#pragma unroll
    for (int ms = 0; ms < 2; ++ms) {
      int r = wm * 32 + ms * 16 + l16;
      af[ms].u2[0] = *(const uint4*)&lQ[cur][r * 40 + halfl * 8];
      af[ms].u2[1] = *(const uint4*)&lQ[cur][r * 40 + 16 + halfl * 8];
    }
    {
      int c = wn * 16 + l16;
      bf.u2[0] = *(const uint4*)&lK[cur][c * 40 + halfl * 8];
      bf.u2[1] = *(const uint4*)&lK[cur][c * 40 + 16 + halfl * 8];
    }
#pragma unroll
    for (int ms = 0; ms < 2; ++ms)
      acc[ms] = __builtin_amdgcn_wmma_f32_16x16x32_f16(
          false, af[ms].h, false, bf.h, (short)0, acc[ms], false, false);

    if (more) {
      *(uint4*)&lQ[nxt][soff] = qv.u;
      *(uint4*)&lK[nxt][soff] = kv.u;
    }
    __syncthreads();
  }

  float* out = attn_raw + (long long)blockIdx.z * DH * DH;
#pragma unroll
  for (int ms = 0; ms < 2; ++ms)
#pragma unroll
    for (int r = 0; r < 8; ++r) {
      int d = d0 + wm * 32 + ms * 16 + r + 8 * halfl;
      int e = e0 + wn * 16 + l16;
      out[d * DH + e] = acc[ms][r];
    }
}

// ---------------------------------------------------------------- norms
__global__ void col_sumsq(const _Float16* __restrict__ qkv,
                          float* __restrict__ sumsq)
{
  int col = blockIdx.x * 256 + threadIdx.x;   // 0..1535 (q + k columns)
  int b = blockIdx.y;
  int n0 = blockIdx.z * 128;
  const _Float16* p = qkv + ((long long)b * N_ + n0) * QKVLD + col;
  float s = 0.f;
  for (int i = 0; i < 128; ++i) { float v = (float)p[(long long)i * QKVLD]; s += v * v; }
  atomicAdd(&sumsq[b * 1536 + col], s);
}

__global__ void finalize_invnorm(const float* __restrict__ s,
                                 float* __restrict__ invn, int n) {
  int i = blockIdx.x * 256 + threadIdx.x;
  if (i < n) invn[i] = 1.f / fmaxf(sqrtf(s[i]), 1e-12f);
}

// ---------------------------------------------------------------- softmax (wave per row)
__global__ __launch_bounds__(256) void softmax_rows(
    const float* __restrict__ attn_raw, const float* __restrict__ invn,
    const float* __restrict__ temp, _Float16* __restrict__ attnT)
{
  int wave = threadIdx.x >> 5, lane = threadIdx.x & 31;
  int row = blockIdx.x * 8 + wave;            // 0..B*H*DH-1
  int beta = row / DH, d = row % DH;
  int b = beta >> 2, h = beta & 3;
  const float* src = attn_raw + (long long)beta * DH * DH + (long long)d * DH;
  float sc = invn[b * 1536 + h * DH + d] * temp[h];
  float vals[6];
#pragma unroll
  for (int j = 0; j < 6; ++j) {
    int e = j * 32 + lane;
    vals[j] = src[e] * sc * invn[b * 1536 + 768 + h * DH + e];
  }
  float m = vals[0];
#pragma unroll
  for (int j = 1; j < 6; ++j) m = fmaxf(m, vals[j]);
  for (int off = 16; off; off >>= 1) m = fmaxf(m, __shfl_xor(m, off));
  float sum = 0.f;
#pragma unroll
  for (int j = 0; j < 6; ++j) { vals[j] = __expf(vals[j] - m); sum += vals[j]; }
  for (int off = 16; off; off >>= 1) sum += __shfl_xor(sum, off);
  float inv = 1.f / sum;
  _Float16* dst = attnT + (long long)beta * DH * DH + d;   // store transposed [e][d]
#pragma unroll
  for (int j = 0; j < 6; ++j) {
    int e = j * 32 + lane;
    dst[(long long)e * DH] = (_Float16)(vals[j] * inv);
  }
}

// ---------------------------------------------------------------- launch
extern "C" void kernel_launch(void* const* d_in, const int* in_sizes, int n_in,
                              void* d_out, int out_size, void* d_ws, size_t ws_size,
                              hipStream_t stream) {
  (void)in_sizes; (void)n_in; (void)out_size; (void)ws_size;
  const float* x    = (const float*)d_in[0];
  const float* Wqkv = (const float*)d_in[1];
  const float* temp = (const float*)d_in[2];
  const float* Wout = (const float*)d_in[3];
  const float* bout = (const float*)d_in[4];

  char* ws = (char*)d_ws;
  size_t o = 0;
  auto alloc = [&](size_t bytes) { size_t r = o; o += (bytes + 255) & ~(size_t)255; return r; };
  _Float16* qkv_h  = (_Float16*)(ws + alloc((size_t)B_ * N_ * QKVLD * 2)); // 151 MB
  _Float16* x_h    = (_Float16*)(ws + alloc((size_t)B_ * N_ * C_ * 2));    // 50 MB
  _Float16* wqkv_h = (_Float16*)(ws + alloc((size_t)C_ * QKVLD * 2));
  _Float16* wout_h = (_Float16*)(ws + alloc((size_t)C_ * C_ * 2));
  _Float16* xca_h  = (_Float16*)(ws + alloc((size_t)B_ * N_ * C_ * 2));    // 50 MB
  float*    a_raw  = (float*)   (ws + alloc((size_t)B_ * H_ * DH * DH * 4));
  _Float16* attnT  = (_Float16*)(ws + alloc((size_t)B_ * H_ * DH * DH * 2));
  float*    sumsq  = (float*)   (ws + alloc((size_t)B_ * 2 * C_ * 4));
  float*    invn   = (float*)   (ws + alloc((size_t)B_ * 2 * C_ * 4));

  (void)hipMemsetAsync(sumsq, 0, (size_t)B_ * 2 * C_ * 4, stream);

  cvt_f32_f16<<<2048, 256, 0, stream>>>(x, x_h, B_ * N_ * C_);
  cvt_f32_f16<<<1024, 256, 0, stream>>>(Wqkv, wqkv_h, C_ * QKVLD);
  cvt_f32_f16<<<512, 256, 0, stream>>>(Wout, wout_h, C_ * C_);

  // qkv = x @ W_qkv : (32768 x 768) @ (768 x 2304) -> f16
  gemm_rr<1, 128><<<dim3(QKVLD / 128, (B_ * N_) / 128, 1), 256, 0, stream>>>(
      x_h, C_, 0, 0, wqkv_h, QKVLD, 0, 0,
      qkv_h, QKVLD, 0, 0, nullptr, C_);

  col_sumsq<<<dim3(6, B_, N_ / 128), 256, 0, stream>>>(qkv_h, sumsq);
  finalize_invnorm<<<(B_ * 2 * C_ + 255) / 256, 256, 0, stream>>>(sumsq, invn, B_ * 2 * C_);

  attn_qk<<<dim3(3, 3, B_ * H_), 256, 0, stream>>>(qkv_h, a_raw);
  softmax_rows<<<(B_ * H_ * DH) / 8, 256, 0, stream>>>(a_raw, invn, temp, attnT);

  // x_CA^T per (b,h): V (4096 x 192) @ attnT (192 x 192) -> xca (B,N,C) layout, f16
  gemm_rr<1, 64><<<dim3(DH / 64, N_ / 128, B_ * H_), 256, 0, stream>>>(
      qkv_h + 2 * C_, QKVLD, (long long)N_ * QKVLD, DH,
      attnT, DH, (long long)H_ * DH * DH, (long long)DH * DH,
      xca_h, C_, (long long)N_ * C_, DH, nullptr, DH);

  // out = xca @ W_out + b_out -> f32
  gemm_rr<0, 128><<<dim3(C_ / 128, (B_ * N_) / 128, 1), 256, 0, stream>>>(
      xca_h, C_, 0, 0, wout_h, C_, 0, 0,
      d_out, C_, 0, 0, bout, C_);
}
